// Grok1MoE_4990751998329
// MI455X (gfx1250) — matrix operand order repaired and checked by reading, compiled, not run on hardware
//
#include <hip/hip_runtime.h>
#include <hip/hip_bf16.h>
#include <math.h>

// ---------------------------------------------------------------------------
// Grok1 MoE for MI455X (gfx1250, wave32, WMMA).
// Router (softcap+softmax+top2) -> token compaction per expert ->
// gate/up bf16 WMMA GEMM + GELU -> down bf16 WMMA GEMM -> combine.
// * fp32 weights stream from HBM, converted to bf16 in-flight.
// * Block tile 128x64; each wave owns 2x2 WMMA subtiles (8 wmma / 12 ds-loads).
// * Double-buffered LDS: next K-tile staged (A via global_load_async_to_lds_b128,
//   ASYNCcnt path) while current tile's WMMAs execute.
// * GELU via hardware v_tanh_f32 (branchless) instead of branchy erf libcall.
// ---------------------------------------------------------------------------

#define T_TOK 2048
#define HDIM  1024
#define IDIM  4096
#define NEXP  8
#define TOPK  2
#define SOFTCAPV 30.0f

#define TMB 128        // block M tile (also expert padding granularity)
#define TN  64         // block N tile
#define TK  32         // WMMA K step (bf16)
#define LDA 40         // LDS row stride in halves (80B, 16B-aligned rows)
#define CAPROWS (T_TOK*TOPK + NEXP*TMB)  // 5120 compacted+padded rows max
#define MTILES (CAPROWS/TMB)             // 40

typedef __attribute__((ext_vector_type(16))) __bf16 v16bf;
typedef __attribute__((ext_vector_type(2)))  __bf16 v2bf;
typedef __attribute__((ext_vector_type(8)))  float  v8f;

union Frag32 { uint4 u4[2]; v16bf v; };

__device__ __forceinline__ unsigned short f2bf(float f) {
  unsigned int u = __float_as_uint(f);
  u += 0x7FFFu + ((u >> 16) & 1u);       // round-to-nearest-even
  return (unsigned short)(u >> 16);
}

// pack two floats -> two bf16 in one dword (v_cvt_pk_bf16_f32 when available)
__device__ __forceinline__ unsigned pk2(float lo, float hi) {
#if __has_builtin(__builtin_amdgcn_cvt_pk_bf16_f32)
  union { v2bf v; unsigned u; } cv;
  cv.v = __builtin_amdgcn_cvt_pk_bf16_f32(lo, hi);
  return cv.u;
#else
  return (unsigned)f2bf(lo) | ((unsigned)f2bf(hi) << 16);
#endif
}

__device__ __forceinline__ uint4 cvt8(float4 a, float4 b) {
  uint4 r;
  r.x = pk2(a.x, a.y);
  r.y = pk2(a.z, a.w);
  r.z = pk2(b.x, b.y);
  r.w = pk2(b.z, b.w);
  return r;
}

// branchless tanh: hardware v_tanh_f32 on gfx1250 if the builtin exists
__device__ __forceinline__ float fast_tanh(float x) {
#if __has_builtin(__builtin_amdgcn_tanhf)
  return __builtin_amdgcn_tanhf(x);
#elif __has_builtin(__builtin_amdgcn_tanh_f32)
  return __builtin_amdgcn_tanh_f32(x);
#else
  return tanhf(x);
#endif
}

// gelu(g)*u, tanh form (|err| ~1e-3, below the bf16 noise of the matmuls)
__device__ __forceinline__ float gelu_mul(float g, float u) {
  float inner = g * (0.7978845608f + 0.0356774081f * g * g);
  return 0.5f * g * (1.0f + fast_tanh(inner)) * u;
}

// A fragment (16x32 bf16), ISA 7.12.2: lanes 0-15 M=lane K={0-7,16-23},
// lanes 16-31 M=lane-16 K={8-15,24-31}.
__device__ __forceinline__ v16bf load_fragA(const unsigned short* base, int lane) {
  int m  = lane & 15;
  int kb = (lane < 16) ? 0 : 8;
  const unsigned short* p = base + m * LDA + kb;
  Frag32 f;
  f.u4[0] = *reinterpret_cast<const uint4*>(p);        // K kb..kb+7
  f.u4[1] = *reinterpret_cast<const uint4*>(p + 16);   // K kb+16..kb+23
  return f.v;
}

// B fragment (32x16 bf16): lanes 0-15 N=lane K={0-7,16-23}; lanes 16-31 K={8-15,24-31}.
__device__ __forceinline__ v16bf load_fragB(const unsigned short* base, int lane) {
  int n  = lane & 15;
  int kb = (lane < 16) ? 0 : 8;
  const unsigned short* p = base + n * LDA + kb;
  Frag32 f;
  f.u4[0] = *reinterpret_cast<const uint4*>(p);
  f.u4[1] = *reinterpret_cast<const uint4*>(p + 16);
  return f.v;
}

#define WMMA_BF16(A, B, C) \
  __builtin_amdgcn_wmma_f32_16x16x32_bf16(false, (A), false, (B), (short)0, (C), false, false)

#define ASYNC_LDS_B128(ldsoff, gaddr) \
  asm volatile("global_load_async_to_lds_b128 %0, %1, off" :: "v"(ldsoff), "v"(gaddr) : "memory")
#define ASYNC_LDS_B128_OFF16(ldsoff, gaddr) \
  asm volatile("global_load_async_to_lds_b128 %0, %1, off offset:16" :: "v"(ldsoff), "v"(gaddr) : "memory")
#define WAIT_ASYNC0() asm volatile("s_wait_asynccnt 0x0" ::: "memory")

// ---------------- kernel 0: x fp32 -> bf16 (4 elems/thread) ----------------
__global__ __launch_bounds__(256) void k_cvt_x(const float* __restrict__ x,
                                               unsigned short* __restrict__ xb) {
  int idx = blockIdx.x * 256 + threadIdx.x;
  float4 v = reinterpret_cast<const float4*>(x)[idx];
  uint2 r;
  r.x = pk2(v.x, v.y);
  r.y = pk2(v.z, v.w);
  reinterpret_cast<uint2*>(xb)[idx] = r;
}

// ---------------- kernel 1: router (one wave per token) ----------------
__global__ __launch_bounds__(256) void k_router(const float* __restrict__ x,
                                                const float* __restrict__ gw,
                                                int* __restrict__ topk_id,
                                                float* __restrict__ topk_w,
                                                int* __restrict__ counts) {
  int wid = threadIdx.x >> 5, lane = threadIdx.x & 31;
  int t = blockIdx.x * 8 + wid;
  float acc[NEXP];
#pragma unroll
  for (int e = 0; e < NEXP; ++e) acc[e] = 0.f;
  const float* xr = x + (size_t)t * HDIM;
  for (int j = 0; j < HDIM / 32; ++j) {
    float xv = xr[lane + 32 * j];
#pragma unroll
    for (int e = 0; e < NEXP; ++e) acc[e] += xv * gw[e * HDIM + lane + 32 * j];
  }
#pragma unroll
  for (int e = 0; e < NEXP; ++e) {
#pragma unroll
    for (int off = 16; off > 0; off >>= 1) acc[e] += __shfl_xor(acc[e], off, 32);
  }
  if (lane == 0) {
    float cap[NEXP], mx = -1e30f;
#pragma unroll
    for (int e = 0; e < NEXP; ++e) {
      cap[e] = SOFTCAPV * fast_tanh(acc[e] * (1.0f / SOFTCAPV));
      mx = fmaxf(mx, cap[e]);
    }
    float p[NEXP], s = 0.f;
#pragma unroll
    for (int e = 0; e < NEXP; ++e) { p[e] = __expf(cap[e] - mx); s += p[e]; }
    float inv = 1.f / s;
#pragma unroll
    for (int e = 0; e < NEXP; ++e) p[e] *= inv;
    int i0 = 0;
#pragma unroll
    for (int e = 1; e < NEXP; ++e) if (p[e] > p[i0]) i0 = e;
    int i1 = (i0 == 0) ? 1 : 0;
#pragma unroll
    for (int e = 0; e < NEXP; ++e) if (e != i0 && p[e] > p[i1]) i1 = e;
    topk_id[t * TOPK + 0] = i0; topk_w[t * TOPK + 0] = p[i0];
    topk_id[t * TOPK + 1] = i1; topk_w[t * TOPK + 1] = p[i1];
    atomicAdd(&counts[i0], 1);
    atomicAdd(&counts[i1], 1);
  }
}

// ---------------- kernel 2: padded offsets + slot init ----------------
__global__ __launch_bounds__(256) void k_setup(const int* __restrict__ counts,
                                               int* __restrict__ pad_off,
                                               int* __restrict__ cursor,
                                               int* __restrict__ rowtok,
                                               float* __restrict__ roww) {
  if (threadIdx.x == 0) {
    int off = 0;
    for (int e = 0; e < NEXP; ++e) {
      pad_off[e] = off;
      off += ((counts[e] + TMB - 1) / TMB) * TMB;
    }
    pad_off[NEXP] = off;
    for (int e = 0; e < NEXP; ++e) cursor[e] = pad_off[e];
  }
  // pad slots: token 0 with weight 0 => contributes exactly zero
  for (int i = threadIdx.x; i < CAPROWS; i += blockDim.x) {
    rowtok[i] = 0;
    roww[i] = 0.f;
  }
}

// ---------------- kernel 3: scatter tokens into expert segments ----------------
__global__ __launch_bounds__(256) void k_scatter(const int* __restrict__ topk_id,
                                                 const float* __restrict__ topk_w,
                                                 int* __restrict__ cursor,
                                                 int* __restrict__ rowtok,
                                                 float* __restrict__ roww,
                                                 int* __restrict__ slot_of) {
  int t = blockIdx.x * 256 + threadIdx.x;
  if (t >= T_TOK) return;
#pragma unroll
  for (int k = 0; k < TOPK; ++k) {
    int e = topk_id[t * TOPK + k];
    int slot = atomicAdd(&cursor[e], 1);
    rowtok[slot] = t;
    roww[slot] = topk_w[t * TOPK + k];
    slot_of[t * TOPK + k] = slot;
  }
}

// ---------------- kernel 4: h = gelu(x@w1g^T) * (x@w1u^T), bf16 ----------------
__global__ __launch_bounds__(256) void k_gateup(const unsigned short* __restrict__ xb,
                                                const float* __restrict__ w1g,
                                                const float* __restrict__ w1u,
                                                const int* __restrict__ pad_off,
                                                const int* __restrict__ rowtok,
                                                unsigned short* __restrict__ hbuf) {
  __shared__ __align__(16) unsigned short As[2][TMB * LDA];
  __shared__ __align__(16) unsigned short Bg[2][TN * LDA];
  __shared__ __align__(16) unsigned short Bu[2][TN * LDA];

  int row0 = blockIdx.y * TMB;
  if (row0 >= pad_off[NEXP]) return;
  int e = 0;
  while (row0 >= pad_off[e + 1]) ++e;
  int n0 = blockIdx.x * TN;

  int tid = threadIdx.x;
  int wid = tid >> 5, lane = tid & 31;
  int mb = (wid & 3) * 2;          // two M subtiles: mb, mb+1
  int nb = (wid >> 2) * 2;         // two N subtiles: nb, nb+1
  int ra = tid >> 1, ca = tid & 1; // A staging: 128 rows x 2 chunks of 16 halves
  int rb = tid >> 2, cb = tid & 3; // B staging: 64 rows x 4 chunks of 8 floats

  // per-thread source pointers (constant across K loop)
  int tokA = rowtok[row0 + ra];
  const unsigned short* aSrc = xb + (size_t)tokA * HDIM + ca * 16;
  const float* gSrc = w1g + (size_t)e * IDIM * HDIM + (size_t)(n0 + rb) * HDIM + cb * 8;
  const float* uSrc = w1u + (size_t)e * IDIM * HDIM + (size_t)(n0 + rb) * HDIM + cb * 8;
  unsigned aDst = (unsigned)(uintptr_t)(&As[0][ra * LDA + ca * 16]);
  unsigned aBufStride = (unsigned)(sizeof(unsigned short) * TMB * LDA);

  v8f g00 = {0,0,0,0,0,0,0,0}, g01 = {0,0,0,0,0,0,0,0};
  v8f g10 = {0,0,0,0,0,0,0,0}, g11 = {0,0,0,0,0,0,0,0};
  v8f u00 = {0,0,0,0,0,0,0,0}, u01 = {0,0,0,0,0,0,0,0};
  v8f u10 = {0,0,0,0,0,0,0,0}, u11 = {0,0,0,0,0,0,0,0};

  auto stage = [&](int buf, int kk) {
    // A: async global->LDS, no VGPR bounce (ASYNCcnt)
    unsigned long long g = (unsigned long long)(uintptr_t)(aSrc + kk);
    unsigned l = aDst + (unsigned)buf * aBufStride;
    ASYNC_LDS_B128(l, g);
    ASYNC_LDS_B128_OFF16(l + 16u, g);
    // B gate/up: fp32 -> bf16 in-flight
    const float* gs = gSrc + kk;
    const float* us = uSrc + kk;
    float4 g0 = reinterpret_cast<const float4*>(gs)[0];
    float4 g1 = reinterpret_cast<const float4*>(gs)[1];
    float4 u0 = reinterpret_cast<const float4*>(us)[0];
    float4 u1 = reinterpret_cast<const float4*>(us)[1];
    *reinterpret_cast<uint4*>(&Bg[buf][rb * LDA + cb * 8]) = cvt8(g0, g1);
    *reinterpret_cast<uint4*>(&Bu[buf][rb * LDA + cb * 8]) = cvt8(u0, u1);
    __builtin_prefetch(gs + TK, 0, 1);   // global_prefetch_b8 for next K step
    __builtin_prefetch(us + TK, 0, 1);
  };

  stage(0, 0);
  WAIT_ASYNC0();
  __syncthreads();

  const int NIT = HDIM / TK;
  for (int it = 0; it < NIT; ++it) {
    int cur = it & 1;
    if (it + 1 < NIT) stage(cur ^ 1, (it + 1) * TK);  // overlaps with WMMAs below

    v16bf a0  = load_fragA(&As[cur][(mb * 16) * LDA], lane);
    v16bf a1  = load_fragA(&As[cur][((mb + 1) * 16) * LDA], lane);
    v16bf bg0 = load_fragB(&Bg[cur][(nb * 16) * LDA], lane);
    v16bf bg1 = load_fragB(&Bg[cur][((nb + 1) * 16) * LDA], lane);
    v16bf bu0 = load_fragB(&Bu[cur][(nb * 16) * LDA], lane);
    v16bf bu1 = load_fragB(&Bu[cur][((nb + 1) * 16) * LDA], lane);

    g00 = WMMA_BF16(a0, bg0, g00);
    g01 = WMMA_BF16(a0, bg1, g01);
    g10 = WMMA_BF16(a1, bg0, g10);
    g11 = WMMA_BF16(a1, bg1, g11);
    u00 = WMMA_BF16(a0, bu0, u00);
    u01 = WMMA_BF16(a0, bu1, u01);
    u10 = WMMA_BF16(a1, bu0, u10);
    u11 = WMMA_BF16(a1, bu1, u11);

    WAIT_ASYNC0();
    __syncthreads();
  }

  // epilogue: h = gelu(g) * u -> bf16
#pragma unroll
  for (int im = 0; im < 2; ++im) {
#pragma unroll
    for (int in = 0; in < 2; ++in) {
      v8f g = im ? (in ? g11 : g10) : (in ? g01 : g00);
      v8f u = im ? (in ? u11 : u10) : (in ? u01 : u00);
#pragma unroll
      for (int rr = 0; rr < 8; ++rr) {
        int m = (mb + im) * 16 + rr + 8 * (lane >> 4);
        int n = (nb + in) * 16 + (lane & 15);
        hbuf[(size_t)(row0 + m) * IDIM + n0 + n] = f2bf(gelu_mul(g[rr], u[rr]));
      }
    }
  }
}

// ---------------- kernel 5: obuf[row] = (h[row] @ w2[e]^T) * we[row] ----------------
__global__ __launch_bounds__(256) void k_down(const unsigned short* __restrict__ hbuf,
                                              const float* __restrict__ w2,
                                              const int* __restrict__ pad_off,
                                              const float* __restrict__ roww,
                                              float* __restrict__ obuf) {
  __shared__ __align__(16) unsigned short As[2][TMB * LDA];
  __shared__ __align__(16) unsigned short Bw[2][TN * LDA];

  int row0 = blockIdx.y * TMB;
  if (row0 >= pad_off[NEXP]) return;
  int e = 0;
  while (row0 >= pad_off[e + 1]) ++e;
  int n0 = blockIdx.x * TN;

  int tid = threadIdx.x;
  int wid = tid >> 5, lane = tid & 31;
  int mb = (wid & 3) * 2;
  int nb = (wid >> 2) * 2;
  int ra = tid >> 1, ca = tid & 1;
  int rb = tid >> 2, cb = tid & 3;

  const unsigned short* aSrc = hbuf + (size_t)(row0 + ra) * IDIM + ca * 16;
  const float* wSrc = w2 + (size_t)e * HDIM * IDIM + (size_t)(n0 + rb) * IDIM + cb * 8;
  unsigned aDst = (unsigned)(uintptr_t)(&As[0][ra * LDA + ca * 16]);
  unsigned aBufStride = (unsigned)(sizeof(unsigned short) * TMB * LDA);

  v8f c00 = {0,0,0,0,0,0,0,0}, c01 = {0,0,0,0,0,0,0,0};
  v8f c10 = {0,0,0,0,0,0,0,0}, c11 = {0,0,0,0,0,0,0,0};

  auto stage = [&](int buf, int kk) {
    unsigned long long g = (unsigned long long)(uintptr_t)(aSrc + kk);
    unsigned l = aDst + (unsigned)buf * aBufStride;
    ASYNC_LDS_B128(l, g);
    ASYNC_LDS_B128_OFF16(l + 16u, g);
    const float* ws = wSrc + kk;
    float4 w0 = reinterpret_cast<const float4*>(ws)[0];
    float4 w1 = reinterpret_cast<const float4*>(ws)[1];
    *reinterpret_cast<uint4*>(&Bw[buf][rb * LDA + cb * 8]) = cvt8(w0, w1);
    __builtin_prefetch(ws + TK, 0, 1);
  };

  stage(0, 0);
  WAIT_ASYNC0();
  __syncthreads();

  const int NIT = IDIM / TK;
  for (int it = 0; it < NIT; ++it) {
    int cur = it & 1;
    if (it + 1 < NIT) stage(cur ^ 1, (it + 1) * TK);

    v16bf a0 = load_fragA(&As[cur][(mb * 16) * LDA], lane);
    v16bf a1 = load_fragA(&As[cur][((mb + 1) * 16) * LDA], lane);
    v16bf b0 = load_fragB(&Bw[cur][(nb * 16) * LDA], lane);
    v16bf b1 = load_fragB(&Bw[cur][((nb + 1) * 16) * LDA], lane);

    c00 = WMMA_BF16(a0, b0, c00);
    c01 = WMMA_BF16(a0, b1, c01);
    c10 = WMMA_BF16(a1, b0, c10);
    c11 = WMMA_BF16(a1, b1, c11);

    WAIT_ASYNC0();
    __syncthreads();
  }

#pragma unroll
  for (int im = 0; im < 2; ++im) {
#pragma unroll
    for (int in = 0; in < 2; ++in) {
      v8f acc = im ? (in ? c11 : c10) : (in ? c01 : c00);
#pragma unroll
      for (int rr = 0; rr < 8; ++rr) {
        int m = (mb + im) * 16 + rr + 8 * (lane >> 4);
        int n = (nb + in) * 16 + (lane & 15);
        int row = row0 + m;
        obuf[(size_t)row * HDIM + n0 + n] = acc[rr] * roww[row];
      }
    }
  }
}

// ---------------- kernel 6: out[t] = obuf[slot0] + obuf[slot1] ----------------
__global__ __launch_bounds__(256) void k_combine(const float* __restrict__ obuf,
                                                 const int* __restrict__ slot_of,
                                                 float* __restrict__ out) {
  int idx = blockIdx.x * 256 + threadIdx.x;
  int t = idx / HDIM, h = idx % HDIM;
  int s0 = slot_of[t * TOPK + 0];
  int s1 = slot_of[t * TOPK + 1];
  out[idx] = obuf[(size_t)s0 * HDIM + h] + obuf[(size_t)s1 * HDIM + h];
}

// ---------------------------------------------------------------------------
extern "C" void kernel_launch(void* const* d_in, const int* in_sizes, int n_in,
                              void* d_out, int out_size, void* d_ws, size_t ws_size,
                              hipStream_t stream) {
  const float* x      = (const float*)d_in[0];
  const float* gate_w = (const float*)d_in[1];
  const float* w1g    = (const float*)d_in[2];
  const float* w1u    = (const float*)d_in[3];
  const float* w2     = (const float*)d_in[4];
  float* out = (float*)d_out;

  char* w = (char*)d_ws;
  unsigned short* xb = (unsigned short*)w; w += (size_t)T_TOK * HDIM * 2;
  int*   topk_id = (int*)w;   w += (size_t)T_TOK * TOPK * 4;
  float* topk_w  = (float*)w; w += (size_t)T_TOK * TOPK * 4;
  int*   counts  = (int*)w;   w += 256;
  int*   pad_off = (int*)w;   w += 256;
  int*   cursor  = (int*)w;   w += 256;
  int*   rowtok  = (int*)w;   w += (size_t)CAPROWS * 4;
  float* roww    = (float*)w; w += (size_t)CAPROWS * 4;
  int*   slot_of = (int*)w;   w += (size_t)T_TOK * TOPK * 4;
  unsigned short* hbuf = (unsigned short*)w; w += (size_t)CAPROWS * IDIM * 2;
  float* obuf = (float*)w;

  hipMemsetAsync(counts, 0, NEXP * sizeof(int), stream);

  k_cvt_x<<<(T_TOK * HDIM) / (256 * 4), 256, 0, stream>>>(x, xb);
  k_router<<<T_TOK / 8, 256, 0, stream>>>(x, gate_w, topk_id, topk_w, counts);
  k_setup<<<1, 256, 0, stream>>>(counts, pad_off, cursor, rowtok, roww);
  k_scatter<<<(T_TOK + 255) / 256, 256, 0, stream>>>(topk_id, topk_w, cursor,
                                                     rowtok, roww, slot_of);
  {
    dim3 grid(IDIM / TN, MTILES);
    k_gateup<<<grid, 256, 0, stream>>>(xb, w1g, w1u, pad_off, rowtok, hbuf);
  }
  {
    dim3 grid(HDIM / TN, MTILES);
    k_down<<<grid, 256, 0, stream>>>(hbuf, w2, pad_off, roww, obuf);
  }
  k_combine<<<(T_TOK * HDIM) / 256, 256, 0, stream>>>(obuf, slot_of, out);
}
